// W_SW_Attention_90211493085495
// MI455X (gfx1250) — matrix operand order repaired and checked by reading, not compiled, not run on hardware
//
#include <hip/hip_runtime.h>
#include <stdint.h>

typedef _Float16 f16;
typedef _Float16 v8h  __attribute__((ext_vector_type(8)));
typedef _Float16 v16h __attribute__((ext_vector_type(16)));
typedef float    v8f  __attribute__((ext_vector_type(8)));

#define WMMA(a, b, c) __builtin_amdgcn_wmma_f32_16x16x32_f16( \
    false, (a), false, (b), (short)0, (c), false, false)

// ---------------------------------------------------------------------------
// Fragment loaders (contiguous k = half*16 + e in both A and B, wave32)
// ---------------------------------------------------------------------------

// A operand, 16x32 f16. base = row_ptr + kchunk + half*16.
// K = half*16 + e : two 16B loads.
static __device__ __forceinline__ v16h load_a_f16(const f16* base) {
    v16h r;
    ((v8h*)&r)[0] = *(const v8h*)(base);
    ((v8h*)&r)[1] = *(const v8h*)(base + 8);
    return r;
}

// Same A layout but source is f32 (emb), converted on the fly.
static __device__ __forceinline__ v16h load_a_f32(const float* base) {
    v16h r;
#pragma unroll
    for (int i = 0; i < 8; ++i) r[i] = (f16)base[i];
#pragma unroll
    for (int i = 0; i < 8; ++i) r[8 + i] = (f16)base[8 + i];
    return r;
}

// B operand, 32x16 f16 stored as B^T (N-major rows over K).
// base = n_row_ptr + kchunk + half*16 ; K = half*16 + e : 16 contiguous halves.
static __device__ __forceinline__ v16h load_b_f16(const f16* base) {
    v16h r;
    ((v8h*)&r)[0] = *(const v8h*)(base);
    ((v8h*)&r)[1] = *(const v8h*)(base + 8);
    return r;
}

// ---------------------------------------------------------------------------
// Weight transpose + f32->f16 convert: dst[n*K + k] = src[k*N + n]
// ---------------------------------------------------------------------------
__global__ void wtrans_kernel(const float* __restrict__ src, f16* __restrict__ dst,
                              int K, int N) {
    int i = blockIdx.x * blockDim.x + threadIdx.x;
    if (i >= N * K) return;
    int n = i / K, k = i - n * K;
    dst[i] = (f16)src[k * N + n];
}

// ---------------------------------------------------------------------------
// QKV projection: per window (64 tokens) x 128-col tile of Wqkv (768 cols).
// Roll(-4, axis=H) + windowing folded into A-row addressing.
// N-tiles align with q/k/v: blockIdx.y 0-1 -> q, 2-3 -> k, 4-5 -> v.
// q,k stored token-major [*,t,32]; v transposed to [*,hd,64] via padded LDS
// so its global stores are 64B-coalesced.
// ---------------------------------------------------------------------------
__global__ __launch_bounds__(256) void qkv_kernel(
    const float* __restrict__ emb, const f16* __restrict__ WT,
    const float* __restrict__ bias,
    f16* __restrict__ qb, f16* __restrict__ kb, f16* __restrict__ vTb) {
    __shared__ __align__(16) f16 vtile[64][130];   // pad->stride 65 dwords

    const int bw  = blockIdx.x;                 // 0..2047  (b*1024 + win)
    const int b   = bw >> 10;
    const int win = bw & 1023;
    const int wr  = win >> 5, wc = win & 31;
    const int n0  = blockIdx.y * 128;
    const int sel = n0 >> 8;                    // block-uniform: 0=q 1=k 2=v

    const int lane = threadIdx.x & 31;
    const int wave = threadIdx.x >> 5;          // 8 waves
    const int mt   = wave >> 1;                 // M tile 0..3
    const int ng   = (wave & 1) * 4;            // N tile group base
    const int lm   = lane & 15;
    const int half = lane >> 4;

    // L2-resident weight panel: prefetch this block's slice
    __builtin_prefetch(WT + (size_t)(n0 + threadIdx.x / 2) * 256, 0, 1);

    // A row = window token (same rows in both lane halves for A operand)
    const int t  = mt * 16 + lm;
    const int pi = t >> 3, pj = t & 7;
    const int grow = (wr * 8 + pi + 4) & 255;   // roll(-SHIFT) folded in
    const int gcol = wc * 8 + pj;
    const float* arow = emb + (((size_t)b * 256 + grow) * 256 + gcol) * 256;

    v8f acc[4] = {};
#pragma unroll
    for (int kk = 0; kk < 8; ++kk) {            // K = 256 in chunks of 32
        v16h a = load_a_f32(arow + kk * 32 + half * 16);
#pragma unroll
        for (int j = 0; j < 4; ++j) {
            int n = n0 + (ng + j) * 16 + lm;
            v16h bf = load_b_f16(WT + (size_t)n * 256 + kk * 32 + half * 16);
            acc[j] = WMMA(a, bf, acc[j]);
        }
    }

    if (sel < 2) {
        // ---- q / k: token-major rows of HD; 32B-contiguous per 16 lanes
        f16* dst = (sel == 0) ? qb : kb;
#pragma unroll
        for (int j = 0; j < 4; ++j) {
            int n   = n0 + (ng + j) * 16 + lm;  // 0..511
            float bv = bias[n];
            int h  = (n >> 5) & 7;
            int hd = n & 31;
            size_t base = (((size_t)b * 1024 + win) * 8 + h) * 2048;
#pragma unroll
            for (int r = 0; r < 8; ++r) {
                int tt = mt * 16 + r + half * 8;        // C/D row mapping
                dst[base + tt * 32 + hd] = (f16)(acc[j][r] + bv);
            }
        }
    } else {
        // ---- v: stage 64x128 tile in LDS, then coalesced transposed stores
#pragma unroll
        for (int j = 0; j < 4; ++j) {
            int nn  = (ng + j) * 16 + lm;       // 0..127 within tile
            float bv = bias[n0 + nn];
#pragma unroll
            for (int r = 0; r < 8; ++r) {
                int tt = mt * 16 + r + half * 8;
                vtile[tt][nn] = (f16)(acc[j][r] + bv);
            }
        }
        __syncthreads();
        // thread i writes 32 contiguous halves (64B) of one vT row
        const int nn    = threadIdx.x >> 1;     // column 0..127
        const int tpart = (threadIdx.x & 1) * 32;
        const int n     = n0 + nn;
        const int h     = (n >> 5) & 7;
        const int hd    = n & 31;
        f16* dst = vTb + (((size_t)b * 1024 + win) * 8 + h) * 2048 + hd * 64 + tpart;
#pragma unroll
        for (int tt = 0; tt < 32; ++tt)
            dst[tt] = vtile[tpart + tt][nn];
    }
}

// ---------------------------------------------------------------------------
// Attention per (b, win, head): sim = q k^T (K=HD=32, one WMMA per tile),
// LDS softmax with rel-pos bias + shift mask, o = pr @ v (K=64, two WMMAs).
// 128 threads = 4 waves; wave w owns sim rows [16w,16w+16).
// ---------------------------------------------------------------------------
__global__ __launch_bounds__(128) void attn_kernel(
    const f16* __restrict__ qb, const f16* __restrict__ kb,
    const f16* __restrict__ vTb, const float* __restrict__ rel,
    f16* __restrict__ ob) {
    __shared__ float sim[64][64];
    __shared__ __align__(16) f16 pr[64][64];

    const int bw  = blockIdx.x;
    const int b   = bw >> 10;
    const int win = bw & 1023;
    const int h   = blockIdx.y;
    const int wr  = win >> 5, wc = win & 31;
    const int lane = threadIdx.x & 31, wave = threadIdx.x >> 5;
    const int lm = lane & 15, half = lane >> 4;
    const size_t base = (((size_t)b * 1024 + win) * 8 + h) * 2048;

    // ---- phase 1: sim = q k^T -------------------------------------------
    {
        const int row = wave * 16 + lm;
        v16h a = load_a_f16(qb + base + row * 32 + half * 16);
#pragma unroll
        for (int j = 0; j < 4; ++j) {
            v16h bf = load_b_f16(kb + base + (j * 16 + lm) * 32 + half * 16);
            v8f c = {};
            c = WMMA(a, bf, c);
#pragma unroll
            for (int r = 0; r < 8; ++r)
                sim[wave * 16 + r + half * 8][j * 16 + lm] = c[r];
        }
    }
    __syncthreads();

    // ---- phase 2: bias + mask + softmax (one thread per query row) ------
    if (threadIdx.x < 64) {
        const int r  = threadIdx.x;
        const int pi = r >> 3, pj = r & 7;
        const float scale = 0.17677669529663687f;   // 32^-0.5
        float mx = -1e30f;
        for (int q = 0; q < 64; ++q) {
            int qi = q >> 3, qj = q & 7;
            bool msk = ((wr == 31) && ((pi < 4) != (qi < 4))) ||
                       ((wc == 31) && ((pj < 4) != (qj < 4)));
            float x = msk ? -1e30f
                          : sim[r][q] * scale +
                            rel[h * 225 + (pi - qi + 7) * 15 + (pj - qj + 7)];
            sim[r][q] = x;
            mx = fmaxf(mx, x);
        }
        float sum = 0.f;
        for (int q = 0; q < 64; ++q) {
            float e = __expf(sim[r][q] - mx);
            sim[r][q] = e;
            sum += e;
        }
        float inv = 1.f / sum;
        for (int q = 0; q < 64; ++q) pr[r][q] = (f16)(sim[r][q] * inv);
    }
    __syncthreads();

    // ---- phase 3: o = pr @ v (64x64 @ 64x32) ----------------------------
    {
        const int row = wave * 16 + lm;
#pragma unroll
        for (int j = 0; j < 2; ++j) {
            v8f c = {};
#pragma unroll
            for (int kk = 0; kk < 2; ++kk) {
                v16h a  = load_a_f16(&pr[row][kk * 32 + half * 16]);
                v16h bf = load_b_f16(vTb + base + (j * 16 + lm) * 64 +
                                     kk * 32 + half * 16);
                c = WMMA(a, bf, c);
            }
#pragma unroll
            for (int r = 0; r < 8; ++r) {
                int tt = wave * 16 + r + half * 8;
                ob[(((size_t)b * 1024 + win) * 64 + tt) * 256 +
                   h * 32 + j * 16 + lm] = (f16)c[r];
            }
        }
    }
}

// ---------------------------------------------------------------------------
// Output projection (K=256, N=256) with unwindow + roll(+4) scatter + bias.
// ---------------------------------------------------------------------------
__global__ __launch_bounds__(256) void oproj_kernel(
    const f16* __restrict__ ob, const f16* __restrict__ WT,
    const float* __restrict__ bo, float* __restrict__ out) {
    const int bw  = blockIdx.x;
    const int b   = bw >> 10;
    const int win = bw & 1023;
    const int wr  = win >> 5, wc = win & 31;
    const int n0  = blockIdx.y * 128;
    const int lane = threadIdx.x & 31, wave = threadIdx.x >> 5;
    const int lm = lane & 15, half = lane >> 4;
    const int mt = wave >> 1, ng = (wave & 1) * 4;

    __builtin_prefetch(WT + (size_t)(n0 + threadIdx.x / 2) * 256, 0, 1);

    const f16* arow = ob + (((size_t)b * 1024 + win) * 64 + mt * 16 + lm) * 256;
    v8f acc[4] = {};
#pragma unroll
    for (int kk = 0; kk < 8; ++kk) {
        v16h a = load_a_f16(arow + kk * 32 + half * 16);
#pragma unroll
        for (int j = 0; j < 4; ++j) {
            int n = n0 + (ng + j) * 16 + lm;
            v16h bf = load_b_f16(WT + (size_t)n * 256 + kk * 32 + half * 16);
            acc[j] = WMMA(a, bf, acc[j]);
        }
    }
#pragma unroll
    for (int j = 0; j < 4; ++j) {
        int n    = n0 + (ng + j) * 16 + lm;
        float bv = bo[n];
#pragma unroll
        for (int r = 0; r < 8; ++r) {
            int t  = mt * 16 + r + half * 8;
            int pi = t >> 3, pj = t & 7;
            int orow = (wr * 8 + pi + 4) & 255;     // unwindow + roll(+SHIFT)
            int ocol = wc * 8 + pj;
            out[(((size_t)b * 256 + orow) * 256 + ocol) * 256 + n] =
                acc[j][r] + bv;
        }
    }
}

// ---------------------------------------------------------------------------
extern "C" void kernel_launch(void* const* d_in, const int* in_sizes, int n_in,
                              void* d_out, int out_size, void* d_ws,
                              size_t ws_size, hipStream_t stream) {
    (void)in_sizes; (void)n_in; (void)out_size; (void)ws_size;
    const float* emb1  = (const float*)d_in[0];
    const float* emb2  = (const float*)d_in[1];
    const float* Wqkv1 = (const float*)d_in[2];
    const float* bqkv1 = (const float*)d_in[3];
    const float* Wqkv2 = (const float*)d_in[4];
    const float* bqkv2 = (const float*)d_in[5];
    const float* rel1  = (const float*)d_in[6];
    const float* rel2  = (const float*)d_in[7];
    const float* Wo1   = (const float*)d_in[8];
    const float* bo1   = (const float*)d_in[9];
    const float* Wo2   = (const float*)d_in[10];
    const float* bo2   = (const float*)d_in[11];
    float* out = (float*)d_out;

    // ---- workspace carve (f16) ----
    char* p = (char*)d_ws;
    auto carve = [&](size_t halves) -> f16* {
        f16* r = (f16*)p;
        p += halves * sizeof(f16);
        p = (char*)(((uintptr_t)p + 255) & ~(uintptr_t)255);
        return r;
    };
    const size_t QKVSZ = (size_t)2 * 1024 * 8 * 64 * 32;  // 33,554,432 halves
    f16* WqkvT1 = carve(768 * 256);
    f16* WqkvT2 = carve(768 * 256);
    f16* WoT1   = carve(256 * 256);
    f16* WoT2   = carve(256 * 256);
    f16* q1  = carve(QKVSZ);  f16* k1 = carve(QKVSZ);  f16* vT1 = carve(QKVSZ);
    f16* q2  = carve(QKVSZ);  f16* k2 = carve(QKVSZ);  f16* vT2 = carve(QKVSZ);
    f16* ob1 = carve(QKVSZ);  f16* ob2 = carve(QKVSZ);

    // ---- weight transpose + convert ----
    wtrans_kernel<<<(768 * 256 + 255) / 256, 256, 0, stream>>>(Wqkv1, WqkvT1, 256, 768);
    wtrans_kernel<<<(768 * 256 + 255) / 256, 256, 0, stream>>>(Wqkv2, WqkvT2, 256, 768);
    wtrans_kernel<<<(256 * 256 + 255) / 256, 256, 0, stream>>>(Wo1, WoT1, 256, 256);
    wtrans_kernel<<<(256 * 256 + 255) / 256, 256, 0, stream>>>(Wo2, WoT2, 256, 256);

    // ---- QKV projections (both streams) ----
    qkv_kernel<<<dim3(2048, 6), 256, 0, stream>>>(emb1, WqkvT1, bqkv1, q1, k1, vT1);
    qkv_kernel<<<dim3(2048, 6), 256, 0, stream>>>(emb2, WqkvT2, bqkv2, q2, k2, vT2);

    // ---- cross attention: o1 = attn(q1,k2,v2,rel1) ; o2 = attn(q2,k1,v1,rel2)
    attn_kernel<<<dim3(2048, 8), 128, 0, stream>>>(q1, k2, vT2, rel1, ob1);
    attn_kernel<<<dim3(2048, 8), 128, 0, stream>>>(q2, k1, vT1, rel2, ob2);

    // ---- output projections + unwindow/roll scatter ----
    oproj_kernel<<<dim3(2048, 2), 256, 0, stream>>>(ob1, WoT1, bo1, out);
    oproj_kernel<<<dim3(2048, 2), 256, 0, stream>>>(ob2, WoT2, bo2,
                                                    out + (size_t)33554432);
}
